// Pyramid_vig3dUnet_2972117369154
// MI455X (gfx1250) — compile-verified, matrix operand order checked by reading, not run on hardware
//
#include <hip/hip_runtime.h>
#include <hip/hip_bf16.h>

// ---------------- problem constants (from reference setup_inputs) ------------
#define BB   2
#define CC   64
#define HH   20
#define WW   48
#define DD   40
#define NN   (HH*WW*DD)      // 38400 voxels per batch
#define NY   600             // pooled candidates per batch (5*12*10)
#define NYP  608             // padded to multiple of 16
#define KNN  27              // top-k neighbors
#define MTOT (BB*NN)         // 76800 rows

typedef _Float16 half8  __attribute__((ext_vector_type(8)));
typedef _Float16 half16 __attribute__((ext_vector_type(16)));
typedef float    f32x8  __attribute__((ext_vector_type(8)));

#define NEG_INF (-3.402823466e38f)

// Load the per-lane A/B operand slice for v_wmma_f32_16x16x32_f16.
// ISA layout (16-bit A 16x32): lane L -> row (L&15), K = [kb..kb+8) and
// [kb+16..kb+24) with kb = 8*(L>>4).  p must point at row*K + kb (+k0).
__device__ __forceinline__ half16 ldpair(const _Float16* p) {
    half8 lo = *(const half8*)(p);
    half8 hi = *(const half8*)(p + 16);
    return __builtin_shufflevector(lo, hi, 0,1,2,3,4,5,6,7,8,9,10,11,12,13,14,15);
}

__device__ __forceinline__ float gelu_exact(float v) {
    return 0.5f * v * (1.0f + erff(v * 0.70710678118654752f));
}

// ---------------- kernel 0: weight cast f32 -> f16 ---------------------------
__global__ void cast_f16_kernel(const float* __restrict__ src,
                                _Float16* __restrict__ dst, int n) {
    int i = blockIdx.x * blockDim.x + threadIdx.x;
    if (i < n) dst[i] = (_Float16)src[i];
}

// ---------------- kernel 1: x [B,C,N] f32 -> xt [B,N,C] f16 ------------------
__global__ void transpose_cast_kernel(const float* __restrict__ x,
                                      _Float16* __restrict__ xt) {
    int n = blockIdx.x * 256 + threadIdx.x;   // coalesced read along N
    int c = blockIdx.y;
    int b = blockIdx.z;
    float v = x[((size_t)b * CC + c) * NN + n];
    xt[((size_t)b * NN + n) * CC + c] = (_Float16)v;
}

// ---------------- generic fused WMMA GEMM ------------------------------------
// D[m, o] = act( A[m,:] . Wt[o,:] + bias[o] ) (+ residual)
// A: [M, K] f16 row-major, Wt: [O, K] f16 row-major (== column-major B operand)
// 128 threads = 4 waves; each wave owns a 16x64 output tile: one A fragment is
// reused across 4 B tiles / 4 independent accumulators (K fully unrolled).
template<int K, int ACT_GELU, int RESID_MODE, int OUT32_MODE, int DO_OUT16>
__global__ void gemm16_kernel(const _Float16* __restrict__ A,
                              const _Float16* __restrict__ Wt,
                              const float*    __restrict__ bias,
                              const float*    __restrict__ resid,
                              float*          __restrict__ out32,
                              _Float16*       __restrict__ out16,
                              int O) {
    const int lane = threadIdx.x & 31;
    const int wave = threadIdx.x >> 5;
    const int row0 = (blockIdx.x * 4 + wave) * 16;  // per-wave row tile
    const int o0   = blockIdx.y * 64;               // 4 col tiles per wave
    const int l16  = lane >> 4;
    const int lm   = lane & 15;

    const _Float16* ap = A  + (size_t)(row0 + lm) * K + l16 * 8;
    const _Float16* bp = Wt + (size_t)(o0   + lm) * K + l16 * 8;

    f32x8 acc[4] = {};
#pragma unroll
    for (int k0 = 0; k0 < K; k0 += 32) {
        half16 av = ldpair(ap + k0);
#pragma unroll
        for (int ot = 0; ot < 4; ++ot) {
            half16 bv = ldpair(bp + (size_t)(ot * 16) * K + k0);
            acc[ot] = __builtin_amdgcn_wmma_f32_16x16x32_f16(
                false, av, false, bv, (short)0, acc[ot], false, false);
        }
    }

    float bz[4];
#pragma unroll
    for (int ot = 0; ot < 4; ++ot) bz[ot] = bias[o0 + ot * 16 + lm];

    const int mbase = row0 + l16 * 8;
#pragma unroll
    for (int r = 0; r < 8; ++r) {
        const int grow = mbase + r;                 // row in [0, B*N)
        const int bb = grow / NN, nn = grow - (grow / NN) * NN;
#pragma unroll
        for (int ot = 0; ot < 4; ++ot) {
            const int ncol = o0 + ot * 16 + lm;
            float v = acc[ot][r] + bz[ot];
            if (ACT_GELU) v = gelu_exact(v);
            if (RESID_MODE == 1)                     // row-major residual [M,O]
                v += resid[(size_t)grow * O + ncol];
            if (RESID_MODE == 2)                     // channel-major [B,O,N]
                v += resid[((size_t)bb * O + ncol) * NN + nn];
            if (OUT32_MODE == 1) out32[(size_t)grow * O + ncol] = v;
            if (OUT32_MODE == 2)                     // transposed -> [B,O,N]
                out32[((size_t)bb * O + ncol) * NN + nn] = v;
            if (DO_OUT16) out16[(size_t)grow * O + ncol] = (_Float16)v;
        }
    }
}

// ---------------- kernel 3: 4^3 avg pool + ||y||^2 + yT16 --------------------
// grid (NYP, B), block 64 (thread == channel)
__global__ void pool_kernel(const float* __restrict__ hT32,
                            _Float16* __restrict__ yT16,
                            float* __restrict__ y2) {
    int m = blockIdx.x, b = blockIdx.y, c = threadIdx.x;
    __shared__ float red[64];
    if (m >= NY) {                           // zero padding rows
        yT16[((size_t)b * NYP + m) * CC + c] = (_Float16)0.0f;
        if (c == 0) y2[b * NYP + m] = 0.0f;
        return;
    }
    int dy = m % 10, wy = (m / 10) % 12, hy = m / 120;
    float s = 0.0f;
    for (int i = 0; i < 4; ++i)
        for (int j = 0; j < 4; ++j)
            for (int k = 0; k < 4; ++k) {
                int n = ((hy * 4 + i) * WW + (wy * 4 + j)) * DD + (dy * 4 + k);
                s += hT32[((size_t)b * NN + n) * CC + c];
            }
    float avg = s * (1.0f / 64.0f);
    yT16[((size_t)b * NYP + m) * CC + c] = (_Float16)avg;
    red[c] = avg * avg;
    __syncthreads();
    for (int off = 32; off > 0; off >>= 1) {
        if (c < off) red[c] += red[c + off];
        __syncthreads();
    }
    if (c == 0) y2[b * NYP + m] = red[0];
}

// ---------------- kernel 4: fused score-WMMA + top-27 + MRConv + concat ------
// block: 128 threads (4 waves) handles 16 rows (voxels); grid (N/16, B).
// y candidate table is staged in LDS via ASYNCcnt-tracked async copy.
__global__ void score_topk_kernel(const _Float16* __restrict__ hT16,
                                  const float*    __restrict__ hT32,
                                  const _Float16* __restrict__ yT16,
                                  const float*    __restrict__ y2,
                                  const float*    __restrict__ rel_pos,
                                  _Float16*       __restrict__ catT16) {
    __shared__ __attribute__((aligned(16))) _Float16 ytab[NYP * CC]; // 76 KB
    __shared__ float ssc[16 * NYP];                                  // 38 KB
    __shared__ float sx2p[16][8];
    __shared__ float sx2[16];

    const int t  = threadIdx.x;
    const int n0 = blockIdx.x * 16;
    const int b  = blockIdx.y;

    // ---- issue async global->LDS copy of the y table (16B per lane/issue) ---
    {
        const _Float16* gsrc = yT16 + (size_t)b * NYP * CC;
        for (int i = t * 8; i < NYP * CC; i += 128 * 8) {
            unsigned laddr = (unsigned)(size_t)(ytab + i);
            unsigned long long gaddr = (unsigned long long)(size_t)(gsrc + i);
            asm volatile("global_load_async_to_lds_b128 %0, %1, off"
                         :: "v"(laddr), "v"(gaddr) : "memory");
        }
    }

    // ---- phase 0 (overlapped with copy): ||xf||^2 per row (fp32) ----
    {
        int row = t >> 3, part = t & 7;
        const float* hp = hT32 + ((size_t)b * NN + n0 + row) * CC + part * 8;
        float s = 0.0f;
#pragma unroll
        for (int i = 0; i < 8; ++i) s += hp[i] * hp[i];
        sx2p[row][part] = s;
    }
    __syncthreads();
    if (t < 16) {
        float a = 0.0f;
#pragma unroll
        for (int j = 0; j < 8; ++j) a += sx2p[t][j];
        sx2[t] = a;
    }
    asm volatile("s_wait_asynccnt 0x0" ::: "memory");   // my async stores done
    __syncthreads();                                    // everyone's done

    // ---- phase 1: score[16][608] via WMMA (K=64 -> 2 independent accs) ----
    const int lane = t & 31;
    const int wave = t >> 5;
    const int l16  = lane >> 4;
    const int lm   = lane & 15;

    const _Float16* abase = hT16 + ((size_t)b * NN + n0 + lm) * CC + l16 * 8;
    half16 a0 = ldpair(abase);
    half16 a1 = ldpair(abase + 32);

    for (int mt = wave; mt < NYP / 16; mt += 4) {
        int m0 = mt * 16;
        const _Float16* bbase = ytab + (m0 + lm) * CC + l16 * 8;   // LDS
        half16 b0 = ldpair(bbase);
        half16 b1 = ldpair(bbase + 32);
        f32x8 acc0 = {}, acc1 = {};
        acc0 = __builtin_amdgcn_wmma_f32_16x16x32_f16(false, a0, false, b0,
                                                      (short)0, acc0, false, false);
        acc1 = __builtin_amdgcn_wmma_f32_16x16x32_f16(false, a1, false, b1,
                                                      (short)0, acc1, false, false);
        const int mcol   = m0 + lm;
        const int mclamp = mcol < NY ? mcol : NY - 1;
        const float y2v  = y2[b * NYP + mcol];
#pragma unroll
        for (int r = 0; r < 8; ++r) {
            int row = r + l16 * 8;
            float rp = rel_pos[(size_t)(n0 + row) * NY + mclamp];
            float sc = 2.0f * (acc0[r] + acc1[r]) - sx2[row] - y2v + rp;
            ssc[row * NYP + mcol] = (mcol < NY) ? sc : NEG_INF;   // branchless
        }
    }
    __syncthreads();

    // ---- phase 2+3: per-row top-27 then channel-wise max over selected y ----
    for (int rr = 0; rr < 4; ++rr) {
        const int row = wave * 4 + rr;
        const int n   = n0 + row;
        float s[NYP / 32];                    // 19 values / lane
#pragma unroll
        for (int j = 0; j < NYP / 32; ++j) s[j] = ssc[row * NYP + lane + 32 * j];

        int idxs[KNN];
        for (int k = 0; k < KNN; ++k) {
            float bv = NEG_INF;
            int   bj = 0;
#pragma unroll
            for (int j = 0; j < NYP / 32; ++j)
                if (s[j] > bv) { bv = s[j]; bj = j; }
            int bm = lane + 32 * bj;
            // wave max-reduce (tie-break: smaller index) -> all lanes agree
            for (int off = 16; off > 0; off >>= 1) {
                float ov = __shfl_xor(bv, off);
                int   om = __shfl_xor(bm, off);
                if (ov > bv || (ov == bv && om < bm)) { bv = ov; bm = om; }
            }
            idxs[k] = bm;
            if ((bm & 31) == lane) {          // owner removes the winner
                int jz = bm >> 5;
#pragma unroll
                for (int j = 0; j < NYP / 32; ++j)
                    if (j == jz) s[j] = NEG_INF;
            }
        }

        // rel[c] = max_k y[c, idx_k] - xf[c];  cat = [xf | rel]  (f16)
        _Float16* cp = catT16 + ((size_t)b * NN + n) * (2 * CC);
#pragma unroll
        for (int h = 0; h < 2; ++h) {
            int c = lane + h * 32;
            float rm = NEG_INF;
#pragma unroll
            for (int k = 0; k < KNN; ++k)
                rm = fmaxf(rm, (float)ytab[idxs[k] * CC + c]);    // LDS gather
            float xfv = hT32[((size_t)b * NN + n) * CC + c];
            cp[c]      = (_Float16)xfv;
            cp[CC + c] = (_Float16)(rm - xfv);
        }
    }
}

// ---------------------------------------------------------------------------
extern "C" void kernel_launch(void* const* d_in, const int* in_sizes, int n_in,
                              void* d_out, int out_size, void* d_ws, size_t ws_size,
                              hipStream_t stream) {
    const float* x       = (const float*)d_in[0];
    const float* fc1_w   = (const float*)d_in[1];
    const float* fc1_b   = (const float*)d_in[2];
    const float* gc_w    = (const float*)d_in[3];
    const float* gc_b    = (const float*)d_in[4];
    const float* fc2_w   = (const float*)d_in[5];
    const float* fc2_b   = (const float*)d_in[6];
    const float* rel_pos = (const float*)d_in[7];
    const float* ffn1_w  = (const float*)d_in[8];
    const float* ffn1_b  = (const float*)d_in[9];
    const float* ffn2_w  = (const float*)d_in[10];
    const float* ffn2_b  = (const float*)d_in[11];
    float* out = (float*)d_out;

    // ---- workspace layout (256B-aligned slabs) ----
    char* ws = (char*)d_ws;
    size_t off = 0;
    auto alloc = [&](size_t bytes) {
        void* p = ws + off;
        off += (bytes + 255) & ~(size_t)255;
        return p;
    };
    _Float16* xt16    = (_Float16*)alloc((size_t)MTOT * CC * 2);
    _Float16* fc1w16  = (_Float16*)alloc(CC * CC * 2);
    _Float16* gcw16   = (_Float16*)alloc(2 * CC * 2 * CC * 2);
    _Float16* fc2w16  = (_Float16*)alloc(CC * 2 * CC * 2);
    _Float16* ffn1w16 = (_Float16*)alloc(4 * CC * CC * 2);
    _Float16* ffn2w16 = (_Float16*)alloc(CC * 4 * CC * 2);
    float*    hT32    = (float*)   alloc((size_t)MTOT * CC * 4);
    _Float16* hT16    = (_Float16*)alloc((size_t)MTOT * CC * 2);
    _Float16* yT16    = (_Float16*)alloc((size_t)BB * NYP * CC * 2);
    float*    y2      = (float*)   alloc((size_t)BB * NYP * 4);
    _Float16* catT16  = (_Float16*)alloc((size_t)MTOT * 2 * CC * 2);
    _Float16* g16     = (_Float16*)alloc((size_t)MTOT * 2 * CC * 2);
    float*    outG32  = (float*)   alloc((size_t)MTOT * CC * 4);
    _Float16* outG16  = (_Float16*)alloc((size_t)MTOT * CC * 2);
    _Float16* f1_16   = (_Float16*)alloc((size_t)MTOT * 4 * CC * 2);
    (void)ws_size; (void)in_sizes; (void)n_in; (void)out_size;

    const int MB = MTOT / 64;   // 1200 blocks of 4 row-tiles

    // ---- weights -> f16 (already [out,in] == WMMA B-operand layout) ----
    cast_f16_kernel<<<(CC*CC + 255)/256, 256, 0, stream>>>(fc1_w, fc1w16, CC*CC);
    cast_f16_kernel<<<(4*CC*CC + 255)/256, 256, 0, stream>>>(gc_w, gcw16, 4*CC*CC);
    cast_f16_kernel<<<(2*CC*CC + 255)/256, 256, 0, stream>>>(fc2_w, fc2w16, 2*CC*CC);
    cast_f16_kernel<<<(4*CC*CC + 255)/256, 256, 0, stream>>>(ffn1_w, ffn1w16, 4*CC*CC);
    cast_f16_kernel<<<(4*CC*CC + 255)/256, 256, 0, stream>>>(ffn2_w, ffn2w16, 4*CC*CC);

    // ---- x [B,C,N] -> xt [B,N,C] f16 ----
    transpose_cast_kernel<<<dim3(NN/256, CC, BB), 256, 0, stream>>>(x, xt16);

    // ---- fc1: h = x*W1 + b1  -> hT32 + hT16 ----
    gemm16_kernel<64,0,0,1,1><<<dim3(MB, 1), 128, 0, stream>>>(
        xt16, fc1w16, fc1_b, nullptr, hT32, hT16, CC);

    // ---- 4^3 avg pool + ||y||^2 ----
    pool_kernel<<<dim3(NYP, BB), 64, 0, stream>>>(hT32, yT16, y2);

    // ---- fused score + top-27 + MRConv + concat ----
    score_topk_kernel<<<dim3(NN/16, BB), 128, 0, stream>>>(
        hT16, hT32, yT16, y2, rel_pos, catT16);

    // ---- gc: g = gelu(cat*Wg + bg) ----
    gemm16_kernel<128,1,0,0,1><<<dim3(MB, 2), 128, 0, stream>>>(
        catT16, gcw16, gc_b, nullptr, nullptr, g16, 2*CC);

    // ---- fc2: outG = g*W2 + b2 + x (residual, channel-major) ----
    gemm16_kernel<128,0,2,1,1><<<dim3(MB, 1), 128, 0, stream>>>(
        g16, fc2w16, fc2_b, x, outG32, outG16, CC);

    // ---- ffn1: f = gelu(outG*Wf1 + bf1) ----
    gemm16_kernel<64,1,0,0,1><<<dim3(MB, 4), 128, 0, stream>>>(
        outG16, ffn1w16, ffn1_b, nullptr, nullptr, f1_16, 4*CC);

    // ---- ffn2: out[b,c,n] = f*Wf2 + bf2 + outG  (transposed store) ----
    gemm16_kernel<256,0,1,2,0><<<dim3(MB, 1), 128, 0, stream>>>(
        f1_16, ffn2w16, ffn2_b, outG32, out, nullptr, CC);
}
